// FasterRCNN_57569741636143
// MI455X (gfx1250) — matrix operand order repaired and checked by reading, not compiled, not run on hardware
//
#include <hip/hip_runtime.h>
#include <math.h>

typedef _Float16 f16;
typedef __attribute__((ext_vector_type(16))) _Float16 v16h;
typedef __attribute__((ext_vector_type(8)))  _Float16 v8h;
typedef __attribute__((ext_vector_type(8)))  float    v8f;

#define IMGF 512.0f
#define POST_NMS 2000
#define NPROP 6768
#define NDET 18000
#define CLIPV 4.135166556742356f

// ---------------------------------------------------------------- helpers
static __device__ __forceinline__ unsigned sortable_f32(float f) {
  unsigned u = __float_as_uint(f);
  return (u & 0x80000000u) ? ~u : (u | 0x80000000u);
}

static __device__ __forceinline__ v8f wmma_f16f32(v16h a, v16h b, v8f c) {
  return __builtin_amdgcn_wmma_f32_16x16x32_f16(false, a, false, b, (short)0, c,
                                                false, false);
}

static __device__ __forceinline__ v16h pack16(v8h lo, v8h hi) {
  v16h r;
#pragma unroll
  for (int h = 0; h < 8; h++) { r[h] = lo[h]; r[h + 8] = hi[h]; }
  return r;
}

// ---------------------------------------------------------------- converts
__global__ void k_f32_to_f16(const float* __restrict__ src, f16* __restrict__ dst, int n) {
  int i = blockIdx.x * blockDim.x + threadIdx.x;
  if (i < n) dst[i] = (f16)src[i];
}

// rpn conv weights, reordered tap-major: Wk2[co][tap*256+ci]
__global__ void k_wconv_reorder(const float* __restrict__ w, f16* __restrict__ Wk2) {
  int i = blockIdx.x * blockDim.x + threadIdx.x;
  if (i >= 256 * 2304) return;
  int co = i / 2304, r = i - co * 2304;
  int tap = r >> 8, ci = r & 255;
  Wk2[i] = (f16)w[(co * 256 + ci) * 9 + tap];
}

// feature map -> zero-padded channel-last f16: Xc[(y+1)*(W+2)+(x+1)][256]
__global__ void k_feat_chlast(const float* __restrict__ f, f16* __restrict__ Xc,
                              int H, int Wd) {
  int Wp = Wd + 2;
  int total = (H + 2) * Wp * 256;
  int i = blockIdx.x * blockDim.x + threadIdx.x;
  if (i >= total) return;
  int pix = i >> 8, ci = i & 255;
  int y = pix / Wp - 1;
  int x = pix - (pix / Wp) * Wp - 1;
  f16 v = (f16)0.f;
  if (y >= 0 && y < H && x >= 0 && x < Wd) v = (f16)f[(size_t)ci * H * Wd + y * Wd + x];
  Xc[i] = v;
}

// ---------------------------------------------------------------- RPN 3x3 conv (implicit GEMM, WMMA)
// Tap-major K (k = tap*256+ci => A offset is chunk*32). One wave: 16(co) x 64(px) strip.
// Software-pipelined: chunk k+1 fragment loads issued before chunk k's WMMAs.
__global__ void __launch_bounds__(256)
k_rpn_conv3x3_wmma(const f16* __restrict__ Wk2, const f16* __restrict__ Xc,
                   const float* __restrict__ bias, float* __restrict__ T,
                   int H, int Wd) {
  const int HW = H * Wd;
  const int Wp = Wd + 2;
  const int lane = threadIdx.x & 31;
  const int wave = blockIdx.x * (blockDim.x >> 5) + (threadIdx.x >> 5);
  const int nquads = HW >> 6;
  const int mt = wave / nquads;
  const int nq = wave - mt * nquads;
  const int ashift = (lane >> 4) << 3;   // A-matrix per-lane K layout: 0 or 8
  const int bshift = (lane >> 4) << 3;   // B-matrix per-lane K layout: 0 or 8
  const f16* arow = Wk2 + (size_t)((mt << 4) + (lane & 15)) * 2304 + ashift;
  int pb[4];
#pragma unroll
  for (int j = 0; j < 4; j++) {
    int n = (nq << 6) + (j << 4) + (lane & 15);
    int py = n / Wd, px = n - py * Wd;
    pb[j] = ((py * Wp + px) << 8) + bshift;  // element offset into padded channel-last feat
  }

  auto loadA = [&](int kk, v16h& A) {
    const f16* p = arow + kk * 32;
    A = pack16(*(const v8h*)p, *(const v8h*)(p + 16));
  };
  auto loadB = [&](int kk, v16h* B) {
    int tap = kk >> 3;
    int dy = tap / 3, dx = tap - dy * 3;
    int off = ((dy * Wp + dx) << 8) + ((kk & 7) << 5);
#pragma unroll
    for (int j = 0; j < 4; j++) {
      const f16* p = Xc + pb[j] + off;
      B[j] = pack16(*(const v8h*)p, *(const v8h*)(p + 16));
    }
  };

  v8f acc[4] = {};
  v16h A0, A1, B0[4], B1[4];
  loadA(0, A0); loadB(0, B0);
  for (int kk = 0; kk < 72; kk += 2) {
    loadA(kk + 1, A1); loadB(kk + 1, B1);
#pragma unroll
    for (int j = 0; j < 4; j++) acc[j] = wmma_f16f32(A0, B0[j], acc[j]);
    int k2 = (kk + 2 < 72) ? kk + 2 : 71;  // tail: harmless redundant reload
    loadA(k2, A0); loadB(k2, B0);
#pragma unroll
    for (int j = 0; j < 4; j++) acc[j] = wmma_f16f32(A1, B1[j], acc[j]);
  }

  const int mbase = (mt << 4) + ((lane >> 4) << 3);
#pragma unroll
  for (int j = 0; j < 4; j++) {
    int n = (nq << 6) + (j << 4) + (lane & 15);
#pragma unroll
    for (int r = 0; r < 8; r++) {
      int mm = mbase + r;
      float v = acc[j][r] + bias[mm];
      T[(size_t)mm * HW + n] = v > 0.f ? v : 0.f;
    }
  }
}

// ---------------------------------------------------------------- FC GEMM (WMMA, LDS A-staging)
// out[m][n] = act( sum_k A[m][k]*Bw[n][k] + bias[n] ), N % 512 == 0, K % 256 == 0, M % 16 == 0.
// block tile 16(M) x 512(N); 8 waves, each 16x64. A staged in LDS as 16x256 super-chunks;
// B fragments double-buffered across the 8 inner 32-k chunks.
__global__ void __launch_bounds__(256)
k_fc_wmma(const f16* __restrict__ A, const f16* __restrict__ Bw,
          const float* __restrict__ bias, float* __restrict__ outF32,
          f16* __restrict__ outF16, int M, int N, int K, int relu) {
  __shared__ f16 Ash[16 * 256];
  const int tid = threadIdx.x;
  const int lane = tid & 31;
  const int wv = tid >> 5;          // 0..7
  const int nblocks = N >> 9;       // N / 512
  const int mt = blockIdx.x / nblocks;
  const int nb = blockIdx.x - mt * nblocks;
  const int m0 = mt << 4;
  const int ashift = (lane >> 4) << 3;
  const int bshift = (lane >> 4) << 3;
  int nnv[4];
  const f16* brv[4];
#pragma unroll
  for (int j = 0; j < 4; j++) {
    nnv[j] = (nb << 9) + (wv << 6) + (j << 4) + (lane & 15);
    brv[j] = Bw + (size_t)nnv[j] * K + bshift;
  }
  const int srow = tid >> 4;          // 0..15 : all 256 threads stage 16 halves each
  const int scol = (tid & 15) << 4;

  auto loadBf = [&](int kabs, v16h* B) {
#pragma unroll
    for (int j = 0; j < 4; j++) {
      const f16* p = brv[j] + kabs;
      B[j] = pack16(*(const v8h*)p, *(const v8h*)(p + 16));
    }
  };
  auto ldsA = [&](int c) -> v16h {
    const f16* p = Ash + (lane & 15) * 256 + (c << 5) + ashift;
    return pack16(*(const v8h*)p, *(const v8h*)(p + 16));
  };

  v8f acc[4] = {};
  v16h B0[4], B1[4];
  for (int kt = 0; kt < K; kt += 256) {
    __syncthreads();
    {
      const f16* src = A + (size_t)(m0 + srow) * K + kt + scol;
      *(v8h*)(Ash + srow * 256 + scol) = *(const v8h*)src;
      *(v8h*)(Ash + srow * 256 + scol + 8) = *(const v8h*)(src + 8);
    }
    __syncthreads();
    loadBf(kt, B0);
#pragma unroll
    for (int c = 0; c < 8; c += 2) {
      loadBf(kt + ((c + 1) << 5), B1);
      v16h a0 = ldsA(c);
#pragma unroll
      for (int j = 0; j < 4; j++) acc[j] = wmma_f16f32(a0, B0[j], acc[j]);
      int cn = (c + 2 < 8) ? c + 2 : 7;  // tail of super-chunk: redundant reload
      loadBf(kt + (cn << 5), B0);
      v16h a1 = ldsA(c + 1);
#pragma unroll
      for (int j = 0; j < 4; j++) acc[j] = wmma_f16f32(a1, B1[j], acc[j]);
    }
  }

  const int mbase = m0 + ((lane >> 4) << 3);
#pragma unroll
  for (int j = 0; j < 4; j++) {
#pragma unroll
    for (int r = 0; r < 8; r++) {
      int mm = mbase + r;
      float v = acc[j][r] + bias[nnv[j]];
      if (relu) v = fmaxf(v, 0.f);
      if (outF32) outF32[(size_t)mm * N + nnv[j]] = v;
      if (outF16) outF16[(size_t)mm * N + nnv[j]] = (f16)v;
    }
  }
}

// ---------------------------------------------------------------- RPN 1x1 heads
__global__ void k_rpn_heads(const float* __restrict__ T, const float* __restrict__ cw,
                            const float* __restrict__ cb, const float* __restrict__ bw,
                            const float* __restrict__ bb, float* __restrict__ obj,
                            float* __restrict__ del, int HW) {
  int p = blockIdx.x * blockDim.x + threadIdx.x;
  if (p >= HW) return;
  float ao[3], ad[12];
#pragma unroll
  for (int a = 0; a < 3; a++) ao[a] = cb[a];
#pragma unroll
  for (int a = 0; a < 12; a++) ad[a] = bb[a];
  for (int ci = 0; ci < 256; ci++) {
    float t = T[(size_t)ci * HW + p];
#pragma unroll
    for (int a = 0; a < 3; a++) ao[a] += cw[a * 256 + ci] * t;
#pragma unroll
    for (int a = 0; a < 12; a++) ad[a] += bw[a * 256 + ci] * t;
  }
#pragma unroll
  for (int a = 0; a < 3; a++) obj[p * 3 + a] = ao[a];
#pragma unroll
  for (int a = 0; a < 3; a++)
#pragma unroll
    for (int c = 0; c < 4; c++) del[(p * 3 + a) * 4 + c] = ad[a * 4 + c];
}

// ---------------------------------------------------------------- anchors + decode
__global__ void k_decode_level(const float* __restrict__ del, float* __restrict__ boxes,
                               int Nl, int Wd, float stride, float asize) {
  int i = blockIdx.x * blockDim.x + threadIdx.x;
  if (i >= Nl) return;
  int p = i / 3, a = i - p * 3;
  const float ratios[3] = {0.5f, 1.0f, 2.0f};
  float hr = sqrtf(ratios[a]), wr = 1.f / hr;
  float wsz = wr * asize, hsz = hr * asize;
  float bx1 = rintf(-wsz * 0.5f), by1 = rintf(-hsz * 0.5f);
  float bx2 = rintf(wsz * 0.5f),  by2 = rintf(hsz * 0.5f);
  float sx = (float)(p % Wd) * stride, sy = (float)(p / Wd) * stride;
  float ax1 = sx + bx1, ay1 = sy + by1, ax2 = sx + bx2, ay2 = sy + by2;
  float w = ax2 - ax1, h = ay2 - ay1;
  float cx = ax1 + 0.5f * w, cy = ay1 + 0.5f * h;
  float dx = del[i * 4 + 0], dy = del[i * 4 + 1];
  float dw = fminf(del[i * 4 + 2], CLIPV), dh = fminf(del[i * 4 + 3], CLIPV);
  float pcx = dx * w + cx, pcy = dy * h + cy;
  float pw = expf(dw) * w, ph = expf(dh) * h;
  boxes[i * 4 + 0] = pcx - 0.5f * pw;
  boxes[i * 4 + 1] = pcy - 0.5f * ph;
  boxes[i * 4 + 2] = pcx + 0.5f * pw;
  boxes[i * 4 + 3] = pcy + 0.5f * ph;
}

// ---------------------------------------------------------------- sort keys / bitonic
__global__ void k_fill_keys(const float* __restrict__ sc, unsigned long long* __restrict__ keys,
                            int n, int npad) {
  int i = blockIdx.x * blockDim.x + threadIdx.x;
  if (i >= npad) return;
  if (i < n) {
    unsigned u = sortable_f32(sc[i]);
    keys[i] = (((unsigned long long)(~u)) << 32) | (unsigned)i;  // asc sort => score desc, idx asc
  } else {
    keys[i] = ~0ull;
  }
}

__global__ void k_bitonic_step(unsigned long long* keys, int n, int k, int j) {
  int i = blockIdx.x * blockDim.x + threadIdx.x;
  if (i >= n) return;
  int ixj = i ^ j;
  if (ixj > i) {
    unsigned long long a = keys[i], b = keys[ixj];
    bool up = ((i & k) == 0);
    if ((a > b) == up) { keys[i] = b; keys[ixj] = a; }
  }
}

// ---------------------------------------------------------------- per-level top-k gather
__global__ void k_gather_topk(const unsigned long long* __restrict__ keys,
                              const float* __restrict__ sc, const float* __restrict__ boxes,
                              int k, float lv, int off, float* __restrict__ csc,
                              float* __restrict__ cbox, float* __restrict__ clvl) {
  int j = blockIdx.x * blockDim.x + threadIdx.x;
  if (j >= k) return;
  unsigned idx = (unsigned)(keys[j] & 0xffffffffu);
  csc[off + j] = sc[idx];
#pragma unroll
  for (int c = 0; c < 4; c++) cbox[(off + j) * 4 + c] = boxes[idx * 4 + c];
  clvl[off + j] = lv;
}

// ---------------------------------------------------------------- RPN proposal prep
__global__ void k_rpn_prep(const float* __restrict__ csc, const float* __restrict__ cbox,
                           const float* __restrict__ clvl, float* __restrict__ psc,
                           float* __restrict__ pbx, float* __restrict__ obox, int n) {
  int i = blockIdx.x * blockDim.x + threadIdx.x;
  if (i >= n) return;
  float s = 1.f / (1.f + expf(-csc[i]));
  float b[4];
#pragma unroll
  for (int c = 0; c < 4; c++) b[c] = fminf(fmaxf(cbox[i * 4 + c], 0.f), IMGF);
  bool valid = ((b[2] - b[0]) >= 1e-3f) && ((b[3] - b[1]) >= 1e-3f);
  s = valid ? s : -1.f;
  psc[i] = s;
  float o = clvl[i] * 2.f * IMGF;
#pragma unroll
  for (int c = 0; c < 4; c++) { pbx[i * 4 + c] = b[c]; obox[i * 4 + c] = b[c] + o; }
}

// ---------------------------------------------------------------- gather sorted boxes for NMS
__global__ void k_gather_sorted(const unsigned long long* __restrict__ keys,
                                const float* __restrict__ obox, float4* __restrict__ sb,
                                float* __restrict__ sarea, int* __restrict__ sidx, int n) {
  int j = blockIdx.x * blockDim.x + threadIdx.x;
  if (j >= n) return;
  int idx = (int)(keys[j] & 0xffffffffu);
  float4 b = make_float4(obox[idx * 4 + 0], obox[idx * 4 + 1],
                         obox[idx * 4 + 2], obox[idx * 4 + 3]);
  sb[j] = b;
  sarea[j] = fmaxf(b.z - b.x, 0.f) * fmaxf(b.w - b.y, 0.f);
  sidx[j] = idx;
}

// ---------------------------------------------------------------- greedy NMS (single block)
__global__ void __launch_bounds__(1024)
k_nms(const float4* __restrict__ sb, const float* __restrict__ sarea,
      unsigned char* __restrict__ keep, int n, float th) {
  extern __shared__ unsigned char kf[];
  for (int t = threadIdx.x; t < n; t += blockDim.x) kf[t] = 1;
  __syncthreads();
  for (int i = 0; i < n - 1; i++) {
    if (kf[i]) {  // uniform (all threads read the same LDS byte)
      float4 bi = sb[i];
      float ai = sarea[i];
      for (int j = i + 1 + threadIdx.x; j < n; j += blockDim.x) {
        float4 bj = sb[j];
        float ix1 = fmaxf(bi.x, bj.x), iy1 = fmaxf(bi.y, bj.y);
        float ix2 = fminf(bi.z, bj.z), iy2 = fminf(bi.w, bj.w);
        float inter = fmaxf(ix2 - ix1, 0.f) * fmaxf(iy2 - iy1, 0.f);
        float iou = inter / (ai + sarea[j] - inter + 1e-12f);
        if (iou > th) kf[j] = 0;
      }
    }
    __syncthreads();
  }
  for (int t = threadIdx.x; t < n; t += blockDim.x) keep[t] = kf[t];
}

// ---------------------------------------------------------------- RPN compaction
__global__ void k_rpn_compact(const unsigned char* __restrict__ keep,
                              const int* __restrict__ sidx, const float* __restrict__ psc,
                              const float* __restrict__ pbx, float* __restrict__ props,
                              int* __restrict__ pvalid, int n) {
  if (threadIdx.x || blockIdx.x) return;
  int cnt = 0;
  for (int j = 0; j < n && cnt < POST_NMS; j++) {
    int i = sidx[j];
    if (keep[j] && psc[i] > 0.f) {
      for (int c = 0; c < 4; c++) props[cnt * 4 + c] = pbx[i * 4 + c];
      pvalid[cnt] = 1;
      cnt++;
    }
  }
  for (; cnt < POST_NMS; cnt++) {
    for (int c = 0; c < 4; c++) props[cnt * 4 + c] = 0.f;
    pvalid[cnt] = 0;
  }
}

// ---------------------------------------------------------------- FPN level assignment
__global__ void k_levels(const float* __restrict__ props, int* __restrict__ lidx) {
  int p = blockIdx.x * blockDim.x + threadIdx.x;
  if (p >= POST_NMS) return;
  const float* b = props + p * 4;
  float area = (b[2] - b[0]) * (b[3] - b[1]);
  float tl = floorf(4.f + log2f(sqrtf(fmaxf(area, 0.f)) / 224.f + 1e-6f));
  float cl = fminf(fmaxf(tl, 2.f), 5.f);
  lidx[p] = (int)cl - 2;
}

// ---------------------------------------------------------------- roi_align (7x7, sr=2)
__global__ void k_roi_align(const float* __restrict__ f0, const float* __restrict__ f1,
                            const float* __restrict__ f2, const float* __restrict__ f3,
                            const float* __restrict__ props, const int* __restrict__ lidx,
                            f16* __restrict__ pooled) {
  int m = blockIdx.x;
  int c = threadIdx.x;
  int l = lidx[m];
  const float* f = (l == 0) ? f0 : (l == 1) ? f1 : (l == 2) ? f2 : f3;
  int H = 128 >> l;
  float scale = 0.25f / (float)(1 << l);
  int HW = H * H;
  const float* pr = props + m * 4;
  float x1 = pr[0] * scale, y1 = pr[1] * scale;
  float x2 = pr[2] * scale, y2 = pr[3] * scale;
  float bw = fmaxf(x2 - x1, 1.f) * (1.f / 7.f);
  float bh = fmaxf(y2 - y1, 1.f) * (1.f / 7.f);
  const float* fc = f + (size_t)c * HW;
  for (int oy = 0; oy < 7; oy++) {
    for (int ox = 0; ox < 7; ox++) {
      float acc = 0.f;
#pragma unroll
      for (int iy = 0; iy < 2; iy++) {
#pragma unroll
        for (int ix = 0; ix < 2; ix++) {
          float y = y1 + ((float)oy + ((float)iy + 0.5f) * 0.5f) * bh;
          float x = x1 + ((float)ox + ((float)ix + 0.5f) * 0.5f) * bw;
          float vw = (y >= -1.f && y <= (float)H && x >= -1.f && x <= (float)H) ? 1.f : 0.f;
          float yc = fminf(fmaxf(y, 0.f), (float)(H - 1));
          float xc = fminf(fmaxf(x, 0.f), (float)(H - 1));
          int yi = (int)floorf(yc), xi = (int)floorf(xc);
          int y2i = min(yi + 1, H - 1), x2i = min(xi + 1, H - 1);
          float ly = yc - (float)yi, lx = xc - (float)xi;
          float hy = 1.f - ly, hx = 1.f - lx;
          acc += vw * (hy * hx * fc[yi * H + xi] + hy * lx * fc[yi * H + x2i] +
                       ly * hx * fc[y2i * H + xi] + ly * lx * fc[y2i * H + x2i]);
        }
      }
      pooled[(size_t)m * 12544 + c * 49 + oy * 7 + ox] = (f16)(acc * 0.25f);
    }
  }
}

// ---------------------------------------------------------------- cls/box heads (small N)
__global__ void k_cls_reg(const float* __restrict__ h7, const float* __restrict__ cw,
                          const float* __restrict__ cb, const float* __restrict__ bw,
                          const float* __restrict__ bb, float* __restrict__ cls,
                          float* __restrict__ reg) {
  int p = blockIdx.x * blockDim.x + threadIdx.x;
  if (p >= POST_NMS) return;
  float ac[10], ar[40];
#pragma unroll
  for (int c = 0; c < 10; c++) ac[c] = 0.f;
#pragma unroll
  for (int r = 0; r < 40; r++) ar[r] = 0.f;
  const float* hp = h7 + (size_t)p * 1024;
  for (int k = 0; k < 1024; k++) {
    float h = hp[k];
#pragma unroll
    for (int c = 0; c < 10; c++) ac[c] += cw[c * 1024 + k] * h;
#pragma unroll
    for (int r = 0; r < 40; r++) ar[r] += bw[r * 1024 + k] * h;
  }
#pragma unroll
  for (int c = 0; c < 10; c++) cls[p * 10 + c] = ac[c] + cb[c];
#pragma unroll
  for (int r = 0; r < 40; r++) reg[p * 40 + r] = ar[r] + bb[r];
}

// ---------------------------------------------------------------- detection post
__global__ void k_det_post(const float* __restrict__ props, const int* __restrict__ pvalid,
                           const float* __restrict__ cls, const float* __restrict__ reg,
                           float* __restrict__ db, float* __restrict__ ds,
                           int* __restrict__ dv, float* __restrict__ dsm,
                           float* __restrict__ dob) {
  int i = blockIdx.x * blockDim.x + threadIdx.x;
  if (i >= NDET) return;
  int p = i / 9;
  int c = i - p * 9 + 1;  // class 1..9
  float mx = -1e30f;
  for (int t = 0; t < 10; t++) mx = fmaxf(mx, cls[p * 10 + t]);
  float sum = 0.f;
  for (int t = 0; t < 10; t++) sum += expf(cls[p * 10 + t] - mx);
  float prob = expf(cls[p * 10 + c] - mx) / sum;
  const float* pr = props + p * 4;
  float w = pr[2] - pr[0], h = pr[3] - pr[1];
  float cx = pr[0] + 0.5f * w, cy = pr[1] + 0.5f * h;
  const float* d = reg + p * 40 + c * 4;
  float dx = d[0] * 0.1f, dy = d[1] * 0.1f;
  float dw = fminf(d[2] * 0.2f, CLIPV), dh = fminf(d[3] * 0.2f, CLIPV);
  float pcx = dx * w + cx, pcy = dy * h + cy;
  float pw = expf(dw) * w, ph = expf(dh) * h;
  float b0 = fminf(fmaxf(pcx - 0.5f * pw, 0.f), IMGF);
  float b1 = fminf(fmaxf(pcy - 0.5f * ph, 0.f), IMGF);
  float b2 = fminf(fmaxf(pcx + 0.5f * pw, 0.f), IMGF);
  float b3 = fminf(fmaxf(pcy + 0.5f * ph, 0.f), IMGF);
  db[i * 4 + 0] = b0; db[i * 4 + 1] = b1; db[i * 4 + 2] = b2; db[i * 4 + 3] = b3;
  ds[i] = prob;
  int v = (pvalid[p] != 0) && (prob > 0.05f) && ((b2 - b0) >= 1e-2f) && ((b3 - b1) >= 1e-2f);
  dv[i] = v;
  dsm[i] = v ? prob : -1.f;
  float o = (float)c * 2.f * IMGF;
  dob[i * 4 + 0] = b0 + o; dob[i * 4 + 1] = b1 + o;
  dob[i * 4 + 2] = b2 + o; dob[i * 4 + 3] = b3 + o;
}

// ---------------------------------------------------------------- final top-100 outputs
__global__ void k_final(const unsigned char* __restrict__ keep, const int* __restrict__ sidx,
                        const int* __restrict__ dv, const float* __restrict__ db,
                        const float* __restrict__ ds, float* __restrict__ out) {
  if (threadIdx.x || blockIdx.x) return;
  int cnt = 0;
  for (int j = 0; j < NDET && cnt < 100; j++) {
    int i = sidx[j];
    if (keep[j] && dv[i]) {
      for (int c = 0; c < 4; c++) out[cnt * 4 + c] = db[i * 4 + c];
      out[400 + cnt] = ds[i];
      out[500 + cnt] = (float)(i % 9 + 1);
      cnt++;
    }
  }
  for (; cnt < 100; cnt++) {
    for (int c = 0; c < 4; c++) out[cnt * 4 + c] = 0.f;
    out[400 + cnt] = 0.f;
    out[500 + cnt] = 0.f;
  }
}

// ================================================================ host side
static void bitonic_sort(unsigned long long* keys, int n, hipStream_t s) {
  int blocks = (n + 255) / 256;
  for (int k = 2; k <= n; k <<= 1)
    for (int j = k >> 1; j > 0; j >>= 1)
      k_bitonic_step<<<blocks, 256, 0, s>>>(keys, n, k, j);
}

extern "C" void kernel_launch(void* const* d_in, const int* in_sizes, int n_in,
                              void* d_out, int out_size, void* d_ws, size_t ws_size,
                              hipStream_t stream) {
  (void)in_sizes; (void)n_in; (void)out_size; (void)ws_size;
  const float* feat[4] = {(const float*)d_in[0], (const float*)d_in[1],
                          (const float*)d_in[2], (const float*)d_in[3]};
  const float* rpn_conv_w = (const float*)d_in[5];
  const float* rpn_conv_b = (const float*)d_in[6];
  const float* rpn_cls_w  = (const float*)d_in[7];
  const float* rpn_cls_b  = (const float*)d_in[8];
  const float* rpn_bbox_w = (const float*)d_in[9];
  const float* rpn_bbox_b = (const float*)d_in[10];
  const float* fc6_w = (const float*)d_in[11];
  const float* fc6_b = (const float*)d_in[12];
  const float* fc7_w = (const float*)d_in[13];
  const float* fc7_b = (const float*)d_in[14];
  const float* cls_w = (const float*)d_in[15];
  const float* cls_b = (const float*)d_in[16];
  const float* box_w = (const float*)d_in[17];
  const float* box_b = (const float*)d_in[18];

  char* ws = (char*)d_ws;
  size_t cur = 0;
  auto take = [&](size_t bytes) -> size_t {
    cur = (cur + 255) & ~(size_t)255;
    size_t o = cur; cur += bytes; return o;
  };
  // persistent region
  size_t o_wk  = take((size_t)256 * 2304 * 2);
  size_t o_w6  = take((size_t)1024 * 12544 * 2);
  size_t o_w7  = take((size_t)1024 * 1024 * 2);
  size_t o_csc = take((size_t)NPROP * 4);
  size_t o_cbx = take((size_t)NPROP * 16);
  size_t o_clv = take((size_t)NPROP * 4);
  size_t o_psc = take((size_t)NPROP * 4);
  size_t o_pbx = take((size_t)NPROP * 16);
  size_t o_obx = take((size_t)NPROP * 16);
  size_t o_k2  = take((size_t)8192 * 8);
  size_t o_sb  = take((size_t)NPROP * 16);
  size_t o_sa  = take((size_t)NPROP * 4);
  size_t o_si  = take((size_t)NPROP * 4);
  size_t o_kp  = take((size_t)NPROP);
  size_t o_pr  = take((size_t)POST_NMS * 16);
  size_t o_pv  = take((size_t)POST_NMS * 4);
  size_t o_li  = take((size_t)POST_NMS * 4);
  size_t arena = (cur + 255) & ~(size_t)255;

  // phase A (RPN) scratch within arena
  size_t ca = arena;
  auto takeA = [&](size_t bytes) -> size_t {
    ca = (ca + 255) & ~(size_t)255;
    size_t o = ca; ca += bytes; return o;
  };
  size_t o_xc  = takeA((size_t)130 * 130 * 256 * 2);  // padded channel-last feat (max level)
  size_t o_T   = takeA((size_t)256 * 16384 * 4);
  size_t o_obj = takeA((size_t)49152 * 4);
  size_t o_del = takeA((size_t)49152 * 16);
  size_t o_dec = takeA((size_t)49152 * 16);
  size_t o_ks  = takeA((size_t)65536 * 8);

  // phase B (detection head) within same arena (reused after proposals exist)
  size_t cb2 = arena;
  auto takeB = [&](size_t bytes) -> size_t {
    cb2 = (cb2 + 255) & ~(size_t)255;
    size_t o = cb2; cb2 += bytes; return o;
  };
  size_t o_pl  = takeB((size_t)2000 * 12544 * 2);
  size_t o_h6  = takeB((size_t)2000 * 1024 * 2);
  size_t o_h7  = takeB((size_t)2000 * 1024 * 4);
  size_t o_cl  = takeB((size_t)2000 * 10 * 4);
  size_t o_rg  = takeB((size_t)2000 * 40 * 4);
  size_t o_db  = takeB((size_t)NDET * 16);
  size_t o_ds  = takeB((size_t)NDET * 4);
  size_t o_dv  = takeB((size_t)NDET * 4);
  size_t o_dm  = takeB((size_t)NDET * 4);
  size_t o_do  = takeB((size_t)NDET * 16);
  size_t o_dk  = takeB((size_t)32768 * 8);
  size_t o_db2 = takeB((size_t)NDET * 16);
  size_t o_da2 = takeB((size_t)NDET * 4);
  size_t o_di2 = takeB((size_t)NDET * 4);
  size_t o_dkp = takeB((size_t)NDET);

  f16* Wk2 = (f16*)(ws + o_wk);
  f16* W6  = (f16*)(ws + o_w6);
  f16* W7  = (f16*)(ws + o_w7);
  f16* Xc  = (f16*)(ws + o_xc);
  float* T = (float*)(ws + o_T);

  // ---- weight conversions ----
  {
    int n = 256 * 2304;
    k_wconv_reorder<<<(n + 255) / 256, 256, 0, stream>>>(rpn_conv_w, Wk2);
    n = 1024 * 12544;
    k_f32_to_f16<<<(n + 255) / 256, 256, 0, stream>>>(fc6_w, W6, n);
    n = 1024 * 1024;
    k_f32_to_f16<<<(n + 255) / 256, 256, 0, stream>>>(fc7_w, W7, n);
  }

  // ---- RPN per level ----
  const int   Hs[4]   = {128, 64, 32, 16};
  const float strd[4] = {4.f, 8.f, 16.f, 32.f};
  const float szs[4]  = {32.f, 64.f, 128.f, 256.f};
  const int   topk[4] = {2000, 2000, 2000, 768};
  const int   koff[4] = {0, 2000, 4000, 6000};
  const int   npad[4] = {65536, 16384, 4096, 1024};
  for (int lv = 0; lv < 4; lv++) {
    int H = Hs[lv], HW = H * H, Nl = HW * 3;
    int tot = (H + 2) * (H + 2) * 256;
    k_feat_chlast<<<(tot + 255) / 256, 256, 0, stream>>>(feat[lv], Xc, H, H);
    k_rpn_conv3x3_wmma<<<HW / 32, 256, 0, stream>>>(Wk2, Xc, rpn_conv_b, T, H, H);
    k_rpn_heads<<<(HW + 255) / 256, 256, 0, stream>>>(
        T, rpn_cls_w, rpn_cls_b, rpn_bbox_w, rpn_bbox_b,
        (float*)(ws + o_obj), (float*)(ws + o_del), HW);
    k_decode_level<<<(Nl + 255) / 256, 256, 0, stream>>>(
        (float*)(ws + o_del), (float*)(ws + o_dec), Nl, H, strd[lv], szs[lv]);
    k_fill_keys<<<(npad[lv] + 255) / 256, 256, 0, stream>>>(
        (float*)(ws + o_obj), (unsigned long long*)(ws + o_ks), Nl, npad[lv]);
    bitonic_sort((unsigned long long*)(ws + o_ks), npad[lv], stream);
    k_gather_topk<<<(topk[lv] + 255) / 256, 256, 0, stream>>>(
        (unsigned long long*)(ws + o_ks), (float*)(ws + o_obj), (float*)(ws + o_dec),
        topk[lv], (float)lv, koff[lv],
        (float*)(ws + o_csc), (float*)(ws + o_cbx), (float*)(ws + o_clv));
  }

  // ---- RPN proposal selection ----
  k_rpn_prep<<<(NPROP + 255) / 256, 256, 0, stream>>>(
      (float*)(ws + o_csc), (float*)(ws + o_cbx), (float*)(ws + o_clv),
      (float*)(ws + o_psc), (float*)(ws + o_pbx), (float*)(ws + o_obx), NPROP);
  k_fill_keys<<<8192 / 256, 256, 0, stream>>>(
      (float*)(ws + o_psc), (unsigned long long*)(ws + o_k2), NPROP, 8192);
  bitonic_sort((unsigned long long*)(ws + o_k2), 8192, stream);
  k_gather_sorted<<<(NPROP + 255) / 256, 256, 0, stream>>>(
      (unsigned long long*)(ws + o_k2), (float*)(ws + o_obx),
      (float4*)(ws + o_sb), (float*)(ws + o_sa), (int*)(ws + o_si), NPROP);
  k_nms<<<1, 1024, NPROP, stream>>>((float4*)(ws + o_sb), (float*)(ws + o_sa),
                                    (unsigned char*)(ws + o_kp), NPROP, 0.7f);
  k_rpn_compact<<<1, 1, 0, stream>>>(
      (unsigned char*)(ws + o_kp), (int*)(ws + o_si), (float*)(ws + o_psc),
      (float*)(ws + o_pbx), (float*)(ws + o_pr), (int*)(ws + o_pv), NPROP);

  // ---- Box head ----
  k_levels<<<(POST_NMS + 255) / 256, 256, 0, stream>>>((float*)(ws + o_pr),
                                                       (int*)(ws + o_li));
  k_roi_align<<<POST_NMS, 256, 0, stream>>>(feat[0], feat[1], feat[2], feat[3],
                                            (float*)(ws + o_pr), (int*)(ws + o_li),
                                            (f16*)(ws + o_pl));
  // fc6: [2000,12544] x [1024,12544]^T -> f16 [2000,1024]; blocks = 125 * (1024/512)
  k_fc_wmma<<<250, 256, 0, stream>>>((f16*)(ws + o_pl), W6, fc6_b, (float*)nullptr,
                                     (f16*)(ws + o_h6), 2000, 1024, 12544, 1);
  // fc7: [2000,1024] x [1024,1024]^T -> f32 [2000,1024]
  k_fc_wmma<<<250, 256, 0, stream>>>((f16*)(ws + o_h6), W7, fc7_b, (float*)(ws + o_h7),
                                     (f16*)nullptr, 2000, 1024, 1024, 1);
  k_cls_reg<<<(POST_NMS + 63) / 64, 64, 0, stream>>>(
      (float*)(ws + o_h7), cls_w, cls_b, box_w, box_b,
      (float*)(ws + o_cl), (float*)(ws + o_rg));

  // ---- detections ----
  k_det_post<<<(NDET + 255) / 256, 256, 0, stream>>>(
      (float*)(ws + o_pr), (int*)(ws + o_pv), (float*)(ws + o_cl), (float*)(ws + o_rg),
      (float*)(ws + o_db), (float*)(ws + o_ds), (int*)(ws + o_dv),
      (float*)(ws + o_dm), (float*)(ws + o_do));
  k_fill_keys<<<32768 / 256, 256, 0, stream>>>(
      (float*)(ws + o_dm), (unsigned long long*)(ws + o_dk), NDET, 32768);
  bitonic_sort((unsigned long long*)(ws + o_dk), 32768, stream);
  k_gather_sorted<<<(NDET + 255) / 256, 256, 0, stream>>>(
      (unsigned long long*)(ws + o_dk), (float*)(ws + o_do),
      (float4*)(ws + o_db2), (float*)(ws + o_da2), (int*)(ws + o_di2), NDET);
  k_nms<<<1, 1024, NDET, stream>>>((float4*)(ws + o_db2), (float*)(ws + o_da2),
                                   (unsigned char*)(ws + o_dkp), NDET, 0.5f);
  k_final<<<1, 1, 0, stream>>>((unsigned char*)(ws + o_dkp), (int*)(ws + o_di2),
                               (int*)(ws + o_dv), (float*)(ws + o_db),
                               (float*)(ws + o_ds), (float*)d_out);
}
